// CAE_7748121002324
// MI455X (gfx1250) — matrix operand order repaired and checked by reading, not compiled, not run on hardware
//
#include <hip/hip_runtime.h>
#include <hip/hip_bf16.h>
#include <math.h>

#define BATCH 16
#define CCH   512
#define KCL   64
#define NPIX  4096
#define STAGES 3

typedef __attribute__((ext_vector_type(16))) __bf16 v16bf;
typedef __attribute__((ext_vector_type(8)))  __bf16 v8bf;
typedef __attribute__((ext_vector_type(4)))  __bf16 v4bf;
typedef __attribute__((ext_vector_type(8)))  float  v8f;

static __device__ __forceinline__ v16bf cat8(v8bf lo, v8bf hi) {
    return __builtin_shufflevector(lo, hi, 0, 1, 2, 3, 4, 5, 6, 7,
                                           8, 9, 10, 11, 12, 13, 14, 15);
}

// ---------------------------------------------------------------------------
// Batched GEMM: C[z] = op(A[z]) * op(B[z]) (+bias)(relu)
//   f32 in global memory, converted to bf16 while staging into LDS,
//   computed with v_wmma_f32_16x16x32_bf16 (f32 accumulation).
// TA: A stored as [Kd][M] row-major (A^T given). TB: B stored as [N][Kd].
// REQUIREMENT (guaranteed by all launches below): M % BM == 0, N % BN == 0,
// Kd % BK == 0 -> no bounds checks anywhere.
// Per-wave tile: 32(M) x 64(N) = 2x4 WMMA 16x16 tiles. BK = 32 = WMMA K.
// LDS is K-contiguous for both operands so each lane's fragment is two
// aligned 16-byte chunks -> ds_load_b128 pairs; staging writes 8-byte v4bf.
// ---------------------------------------------------------------------------
template<int WAVES_M, int WAVES_N, bool TA, bool TB, bool BIAS, bool RELU>
__global__ __launch_bounds__(WAVES_M * WAVES_N * 32)
void gemm_wmma(const float* __restrict__ A, long sA, int lda,
               const float* __restrict__ B, long sB, int ldb,
               float* __restrict__ C, long sC, int ldc,
               const float* __restrict__ bias, int Kd)
{
    constexpr int BM = 32 * WAVES_M;
    constexpr int BN = 64 * WAVES_N;
    constexpr int BK = 32;
    constexpr int KP = BK + 8;          // row stride (elements); 80B, 16B multiple
    constexpr int NT = WAVES_M * WAVES_N * 32;
    constexpr int ACH = BM * BK / 4;    // 4-element chunks in A tile
    constexpr int BCH = BN * BK / 4;

    __shared__ __attribute__((aligned(16))) __bf16 As[BM][KP];   // As[m][k]
    __shared__ __attribute__((aligned(16))) __bf16 Bs[BN][KP];   // Bs[n][k]

    const float* Ab = A + (long)blockIdx.z * sA;
    const float* Bb = B + (long)blockIdx.z * sB;
    float*       Cb = C + (long)blockIdx.z * sC;

    const int m0 = blockIdx.y * BM;
    const int n0 = blockIdx.x * BN;
    const int tid  = threadIdx.x;
    const int lane = tid & 31;
    const int wave = tid >> 5;
    const int wm = (wave % WAVES_M) * 32;
    const int wn = (wave / WAVES_M) * 64;
    const int lr = lane & 15;
    const int hi = lane >> 4;

    v8f acc[2][4] = {};

    for (int k0 = 0; k0 < Kd; k0 += BK) {
        // ---- stage A tile: one v4bf chunk per thread per iteration ----
        #pragma unroll
        for (int it = 0; it < ACH / NT; ++it) {
            int c = tid + it * NT;
            v4bf pk;
            if (TA) {                           // coalesced along M, strided K reads
                int m = c % BM, kq = c / BM;
                const float* g = Ab + (long)(k0 + 4 * kq) * lda + (m0 + m);
                pk[0] = (__bf16)g[0];
                pk[1] = (__bf16)g[(long)lda];
                pk[2] = (__bf16)g[2 * (long)lda];
                pk[3] = (__bf16)g[3 * (long)lda];
                *(v4bf*)&As[m][4 * kq] = pk;
            } else {                            // contiguous float4 along K
                int m = c / (BK / 4), kq = c % (BK / 4);
                float4 f = *(const float4*)(Ab + (long)(m0 + m) * lda + k0 + 4 * kq);
                pk[0] = (__bf16)f.x; pk[1] = (__bf16)f.y;
                pk[2] = (__bf16)f.z; pk[3] = (__bf16)f.w;
                *(v4bf*)&As[m][4 * kq] = pk;
            }
        }
        // ---- stage B tile into Bs[n][k] ----
        #pragma unroll
        for (int it = 0; it < BCH / NT; ++it) {
            int c = tid + it * NT;
            v4bf pk;
            if (TB) {                           // contiguous float4 along K
                int n = c / (BK / 4), kq = c % (BK / 4);
                float4 f = *(const float4*)(Bb + (long)(n0 + n) * ldb + k0 + 4 * kq);
                pk[0] = (__bf16)f.x; pk[1] = (__bf16)f.y;
                pk[2] = (__bf16)f.z; pk[3] = (__bf16)f.w;
                *(v4bf*)&Bs[n][4 * kq] = pk;
            } else {                            // coalesced along N, strided K reads
                int n = c % BN, kq = c / BN;
                const float* g = Bb + (long)(k0 + 4 * kq) * ldb + (n0 + n);
                pk[0] = (__bf16)g[0];
                pk[1] = (__bf16)g[(long)ldb];
                pk[2] = (__bf16)g[2 * (long)ldb];
                pk[3] = (__bf16)g[3 * (long)ldb];
                *(v4bf*)&Bs[n][4 * kq] = pk;
            }
        }
        __syncthreads();

        // ---- gather fragments (ISA 7.12.2 16-bit layouts), 2x b128 each ----
        v16bf afrag[2], bfrag[4];
        const int kb   = hi ? 8 : 0;    // A: lanes<16 K{0..7,16..23}; >=16 K{8..15,24..31}
        const int koff = hi ? 8 : 0;    // B: lanes<16 K{0..7,16..23}; >=16 K{8..15,24..31}
        #pragma unroll
        for (int i = 0; i < 2; ++i) {
            const __bf16* ap = &As[wm + 16 * i + lr][0];
            afrag[i] = cat8(*(const v8bf*)(ap + kb), *(const v8bf*)(ap + 16 + kb));
        }
        #pragma unroll
        for (int j = 0; j < 4; ++j) {
            const __bf16* bp = &Bs[wn + 16 * j + lr][0];
            bfrag[j] = cat8(*(const v8bf*)(bp + koff), *(const v8bf*)(bp + 16 + koff));
        }

        #pragma unroll
        for (int i = 0; i < 2; ++i)
            #pragma unroll
            for (int j = 0; j < 4; ++j)
                acc[i][j] = __builtin_amdgcn_wmma_f32_16x16x32_bf16(
                    false, afrag[i], false, bfrag[j],
                    (short)0, acc[i][j], false, false);

        __syncthreads();
    }

    // ---- epilogue: C layout VGPR r -> M = r (lanes<16) / 8+r (lanes>=16) ----
    #pragma unroll
    for (int i = 0; i < 2; ++i) {
        #pragma unroll
        for (int j = 0; j < 4; ++j) {
            int col = n0 + wn + 16 * j + lr;
            #pragma unroll
            for (int r = 0; r < 8; ++r) {
                int row = m0 + wm + 16 * i + r + (hi ? 8 : 0);
                float v = acc[i][j][r];
                if (BIAS) v += bias[row];
                if (RELU) v = fmaxf(v, 0.0f);
                Cb[(long)row * ldc + col] = v;
            }
        }
    }
}

// ---------------------------------------------------------------------------
__global__ void zero_kernel(float* p, int n) {
    int i = blockIdx.x * blockDim.x + threadIdx.x;
    if (i < n) p[i] = 0.0f;
}

__global__ void bcast_mu(const float* __restrict__ mu_in, float* __restrict__ mu_w) {
    int i = blockIdx.x * blockDim.x + threadIdx.x;
    if (i < BATCH * CCH * KCL) mu_w[i] = mu_in[i % (CCH * KCL)];
}

// softmax over k (64) per row, in place; accumulate column sums per (b,k)
__global__ void softmax_colsum(float* __restrict__ z, float* __restrict__ zsum) {
    __shared__ float cs[KCL];
    int tid = threadIdx.x;
    if (tid < KCL) cs[tid] = 0.0f;
    __syncthreads();
    int wave = tid >> 5, lane = tid & 31;
    long row = (long)blockIdx.x * 8 + wave;            // over BATCH*NPIX
    float* zr = z + row * KCL;
    float v0 = zr[lane], v1 = zr[lane + 32];
    float m = fmaxf(v0, v1);
    #pragma unroll
    for (int off = 16; off; off >>= 1) m = fmaxf(m, __shfl_xor(m, off, 32));
    float e0 = __expf(v0 - m), e1 = __expf(v1 - m);
    float s = e0 + e1;
    #pragma unroll
    for (int off = 16; off; off >>= 1) s += __shfl_xor(s, off, 32);
    float inv = 1.0f / s;
    e0 *= inv; e1 *= inv;
    zr[lane] = e0; zr[lane + 32] = e1;
    atomicAdd(&cs[lane], e0);
    atomicAdd(&cs[lane + 32], e1);
    __syncthreads();
    int b = (int)((blockIdx.x * 8) / NPIX);            // block never straddles b
    if (tid < KCL) atomicAdd(&zsum[b * KCL + tid], cs[tid]);
}

__global__ void znorm(const float* __restrict__ z, const float* __restrict__ zsum,
                      float* __restrict__ zn, long total) {
    long i = (long)blockIdx.x * blockDim.x + threadIdx.x;
    if (i >= total) return;
    int k = (int)(i & (KCL - 1));
    int b = (int)(i / ((long)NPIX * KCL));
    zn[i] = z[i] / (1e-6f + zsum[b * KCL + k]);
}

// l2 normalize mu_next over channel axis; write mu_w (and optionally d_out)
__global__ void l2norm_mu(const float* __restrict__ mu_next,
                          float* __restrict__ mu_w, float* __restrict__ mu_out) {
    __shared__ float red[256];
    int b = blockIdx.x / KCL, k = blockIdx.x % KCL;
    const float* src = mu_next + (long)b * CCH * KCL + k;
    int tid = threadIdx.x;
    float v0 = src[(long)tid * KCL];
    float v1 = src[(long)(tid + 256) * KCL];
    red[tid] = v0 * v0 + v1 * v1;
    __syncthreads();
    for (int s = 128; s; s >>= 1) {
        if (tid < s) red[tid] += red[tid + s];
        __syncthreads();
    }
    float inv = 1.0f / (1e-6f + sqrtf(red[0]));
    float* dst = mu_w + (long)b * CCH * KCL + k;
    dst[(long)tid * KCL] = v0 * inv;
    dst[(long)(tid + 256) * KCL] = v1 * inv;
    if (mu_out) {
        float* o = mu_out + (long)b * CCH * KCL + k;
        o[(long)tid * KCL] = v0 * inv;
        o[(long)(tid + 256) * KCL] = v1 * inv;
    }
}

__global__ void bn_reduce(const float* __restrict__ x2,
                          float* __restrict__ bnsum, float* __restrict__ bnsq) {
    __shared__ float s1[256], s2[256];
    int b = blockIdx.x / CCH, o = blockIdx.x % CCH;
    const float* p = x2 + ((long)b * CCH + o) * NPIX;
    int tid = threadIdx.x;
    float a = 0.0f, q = 0.0f;
    for (int i = tid; i < NPIX; i += 256) { float v = p[i]; a += v; q += v * v; }
    s1[tid] = a; s2[tid] = q;
    __syncthreads();
    for (int s = 128; s; s >>= 1) {
        if (tid < s) { s1[tid] += s1[tid + s]; s2[tid] += s2[tid + s]; }
        __syncthreads();
    }
    if (tid == 0) { atomicAdd(&bnsum[o], s1[0]); atomicAdd(&bnsq[o], s2[0]); }
}

__global__ void bn_finalize(const float* __restrict__ bnsum, const float* __restrict__ bnsq,
                            const float* __restrict__ gamma, const float* __restrict__ beta,
                            float* __restrict__ scale, float* __restrict__ shift) {
    int o = blockIdx.x * blockDim.x + threadIdx.x;
    if (o >= CCH) return;
    const float invN = 1.0f / (float)(BATCH * NPIX);
    float mean = bnsum[o] * invN;
    float var  = bnsq[o] * invN - mean * mean;
    float s = gamma[o] * rsqrtf(var + 1e-5f);
    scale[o] = s;
    shift[o] = beta[o] - mean * s;
}

__global__ void bn_apply(const float* __restrict__ x2, const float* __restrict__ xin,
                         const float* __restrict__ scale, const float* __restrict__ shift,
                         float* __restrict__ out, long total) {
    long i = (long)blockIdx.x * blockDim.x + threadIdx.x;
    if (i >= total) return;
    int o = (int)((i / NPIX) % CCH);
    float v = x2[i] * scale[o] + shift[o] + xin[i];
    out[i] = fmaxf(v, 0.0f);
}

__global__ void transpose_z(const float* __restrict__ z, float* __restrict__ zt) {
    __shared__ float tile[32][33];
    int b = blockIdx.z;
    int n0 = blockIdx.x * 32, k0 = blockIdx.y * 32;
    const float* zb = z  + (long)b * NPIX * KCL;
    float*       zo = zt + (long)b * KCL * NPIX;
    int tx = threadIdx.x, ty = threadIdx.y;            // 32 x 8
    #pragma unroll
    for (int r = 0; r < 32; r += 8)
        tile[ty + r][tx] = zb[(long)(n0 + ty + r) * KCL + k0 + tx];
    __syncthreads();
    #pragma unroll
    for (int r = 0; r < 32; r += 8)
        zo[(long)(k0 + ty + r) * NPIX + n0 + tx] = tile[tx][ty + r];
}

// ---------------------------------------------------------------------------
extern "C" void kernel_launch(void* const* d_in, const int* in_sizes, int n_in,
                              void* d_out, int out_size, void* d_ws, size_t ws_size,
                              hipStream_t stream)
{
    const float* x     = (const float*)d_in[0];
    const float* w1    = (const float*)d_in[1];
    const float* b1    = (const float*)d_in[2];
    const float* w2    = (const float*)d_in[3];
    const float* gamma = (const float*)d_in[4];
    const float* beta  = (const float*)d_in[5];
    const float* mu_in = (const float*)d_in[6];

    float* out_main = (float*)d_out;                          // [16,512,64,64]
    float* out_mu   = out_main + (long)BATCH * CCH * NPIX;    // [16,512,64]
    float* out_zt   = out_mu + (long)BATCH * CCH * KCL;       // [16,64,64,64]

    char* wsp = (char*)d_ws;
    auto alloc = [&](size_t bytes) {
        char* p = wsp;
        wsp += (bytes + 255) & ~(size_t)255;
        return p;
    };
    float* xs    = (float*)alloc((size_t)BATCH * CCH * NPIX * 4);   // conv1 out, reused as xr
    float* x2    = (float*)alloc((size_t)BATCH * CCH * NPIX * 4);
    float* z     = (float*)alloc((size_t)BATCH * NPIX * KCL * 4);
    float* zn    = (float*)alloc((size_t)BATCH * NPIX * KCL * 4);
    float* mu_w  = (float*)alloc((size_t)BATCH * CCH * KCL * 4);
    float* mu_n  = (float*)alloc((size_t)BATCH * CCH * KCL * 4);
    float* zsum  = (float*)alloc((size_t)BATCH * KCL * 4);
    float* bnsum = (float*)alloc((size_t)CCH * 4);
    float* bnsq  = (float*)alloc((size_t)CCH * 4);
    float* scale = (float*)alloc((size_t)CCH * 4);
    float* shift = (float*)alloc((size_t)CCH * 4);

    // broadcast mu to all batches
    bcast_mu<<<(BATCH * CCH * KCL + 255) / 256, 256, 0, stream>>>(mu_in, mu_w);

    // conv1: xs[b] = W1 @ x[b] + b1    (M=512, N=4096, K=512)
    {
        dim3 g(NPIX / 128, CCH / 128, BATCH);
        gemm_wmma<4, 2, false, false, true, false><<<g, 256, 0, stream>>>(
            w1, 0, CCH, x, (long)CCH * NPIX, NPIX,
            xs, (long)CCH * NPIX, NPIX, b1, CCH);
    }

    for (int stage = 0; stage < STAGES; ++stage) {
        // E-step: z[b][n][k] = xs[b]^T @ mu[b]   (M=4096, N=64, K=512)
        {
            dim3 g(1, NPIX / 128, BATCH);
            gemm_wmma<4, 1, true, false, false, false><<<g, 128, 0, stream>>>(
                xs, (long)CCH * NPIX, NPIX, mu_w, (long)CCH * KCL, KCL,
                z, (long)NPIX * KCL, KCL, nullptr, CCH);
        }
        zero_kernel<<<(BATCH * KCL + 255) / 256, 256, 0, stream>>>(zsum, BATCH * KCL);
        softmax_colsum<<<BATCH * NPIX / 8, 256, 0, stream>>>(z, zsum);
        {
            long tot = (long)BATCH * NPIX * KCL;
            znorm<<<(unsigned)((tot + 255) / 256), 256, 0, stream>>>(z, zsum, zn, tot);
        }
        // M-step: mu_n[b] = xs[b] @ zn[b]        (M=512, N=64, K=4096)
        {
            dim3 g(1, CCH / 128, BATCH);
            gemm_wmma<4, 1, false, false, false, false><<<g, 128, 0, stream>>>(
                xs, (long)CCH * NPIX, NPIX, zn, (long)NPIX * KCL, KCL,
                mu_n, (long)CCH * KCL, KCL, nullptr, NPIX);
        }
        l2norm_mu<<<BATCH * KCL, 256, 0, stream>>>(
            mu_n, mu_w, (stage == STAGES - 1) ? out_mu : nullptr);
    }

    // reconstruction: xr[b] = relu(mu[b] @ z[b]^T)   (M=512, N=4096, K=64)
    float* xr = xs;  // xs dead after EM loop
    {
        dim3 g(NPIX / 128, CCH / 128, BATCH);
        gemm_wmma<4, 2, false, true, false, true><<<g, 256, 0, stream>>>(
            mu_w, (long)CCH * KCL, KCL, z, (long)NPIX * KCL, KCL,
            xr, (long)CCH * NPIX, NPIX, nullptr, KCL);
    }
    // conv2: x2[b] = W2 @ xr[b]               (M=512, N=4096, K=512)
    {
        dim3 g(NPIX / 128, CCH / 128, BATCH);
        gemm_wmma<4, 2, false, false, false, false><<<g, 256, 0, stream>>>(
            w2, 0, CCH, xr, (long)CCH * NPIX, NPIX,
            x2, (long)CCH * NPIX, NPIX, nullptr, CCH);
    }
    // BatchNorm (batch stats) + residual + relu
    zero_kernel<<<2, 256, 0, stream>>>(bnsum, CCH);
    zero_kernel<<<2, 256, 0, stream>>>(bnsq, CCH);
    bn_reduce<<<BATCH * CCH, 256, 0, stream>>>(x2, bnsum, bnsq);
    bn_finalize<<<2, 256, 0, stream>>>(bnsum, bnsq, gamma, beta, scale, shift);
    {
        long tot = (long)BATCH * CCH * NPIX;
        bn_apply<<<(unsigned)((tot + 255) / 256), 256, 0, stream>>>(
            x2, x, scale, shift, out_main, tot);
    }
    // z_t output: [b,k,n]
    {
        dim3 g(NPIX / 32, KCL / 32, BATCH);
        transpose_z<<<g, dim3(32, 8), 0, stream>>>(z, out_zt);
    }
}
